// GATGRUModel_20246475833396
// MI455X (gfx1250) — matrix operand order repaired and checked by reading, not compiled, not run on hardware
//
#include <hip/hip_runtime.h>

// ---------------------------------------------------------------- constants
#define B_    1024
#define N_    128
#define F_    16
#define G_    48
#define H_    6
#define D_    8
#define GRU_  256
#define EMB_  128
#define EPSV  1e-5f

typedef _Float16 h16_t;
typedef __attribute__((ext_vector_type(16))) _Float16 v16h;
typedef __attribute__((ext_vector_type(8)))  _Float16 v8h;
typedef __attribute__((ext_vector_type(8)))  float    v8f;

// ---------------------------------------------------------------- wave32 reductions
__device__ __forceinline__ float wred_sum(float v) {
    #pragma unroll
    for (int o = 16; o > 0; o >>= 1) v += __shfl_xor(v, o, 32);
    return v;
}
__device__ __forceinline__ float wred_max(float v) {
    #pragma unroll
    for (int o = 16; o > 0; o >>= 1) v = fmaxf(v, __shfl_xor(v, o, 32));
    return v;
}

// ---------------------------------------------------------------- f32 -> f16 weight converts
__global__ void cvt_f16_kernel(const float* __restrict__ s, h16_t* __restrict__ d, int n) {
    int i = blockIdx.x * 256 + threadIdx.x;
    if (i < n) d[i] = (h16_t)s[i];
}
// src is [K,N] row-major, dst is [N,K] row-major (so GEMM reads contiguous K)
__global__ void cvtT_f16_kernel(const float* __restrict__ s, h16_t* __restrict__ d, int K, int N) {
    int i = blockIdx.x * 256 + threadIdx.x;
    if (i < K * N) {
        int k = i / N, n = i % N;
        d[(size_t)n * K + k] = (h16_t)s[i];
    }
}

// ---------------------------------------------------------------- fused GAT layer
// One block per graph b. X:[B,N,Fin] f32, Adj:[B,N,N] i32, W:[H,Fin,8], a:[H,16].
// out[b,n,head*8+d] = softmax-weighted aggregation (leaky_relu 0.01, mask>0).
template <int Fin>
__global__ __launch_bounds__(256)
void gat_kernel(const float* __restrict__ X, const int* __restrict__ Adj,
                const float* __restrict__ W, const float* __restrict__ a,
                float* __restrict__ out)
{
    __shared__ float Xs[N_ * Fin];
    __shared__ float hs[H_ * N_ * D_];       // 6144 f32
    __shared__ float eis[H_ * N_];
    __shared__ float ejs[H_ * N_];
    __shared__ float Wl[H_ * Fin * D_];
    __shared__ float al[H_ * 2 * D_];

    const int b = blockIdx.x;
    const int tid = threadIdx.x;

    for (int i = tid; i < N_ * Fin; i += 256) Xs[i] = X[(size_t)b * (N_ * Fin) + i];
    for (int i = tid; i < H_ * Fin * D_; i += 256) Wl[i] = W[i];
    if (tid < H_ * 2 * D_) al[tid] = a[tid];
    __syncthreads();

    // h = einsum('nf,hfd->hnd')
    for (int idx = tid; idx < H_ * N_ * D_; idx += 256) {
        int head = idx >> 10;            // / (N_*D_) = /1024
        int rem  = idx & 1023;
        int n = rem >> 3, d = rem & 7;
        const float* xr = &Xs[n * Fin];
        const float* wc = &Wl[head * Fin * D_ + d];
        float acc = 0.f;
        #pragma unroll
        for (int f = 0; f < Fin; ++f) acc += xr[f] * wc[f * D_];
        hs[idx] = acc;
    }
    __syncthreads();

    // per-node src/dst scores
    for (int idx = tid; idx < H_ * N_; idx += 256) {
        int head = idx >> 7, n = idx & 127;
        const float* hv = &hs[head * 1024 + n * 8];
        float s0 = 0.f, s1 = 0.f;
        #pragma unroll
        for (int d = 0; d < 8; ++d) {
            s0 += hv[d] * al[head * 16 + d];
            s1 += hv[d] * al[head * 16 + 8 + d];
        }
        eis[idx] = s0; ejs[idx] = s1;
    }
    __syncthreads();

    // wave-per-(head,row) masked softmax + aggregation
    const int lane = tid & 31, wid = tid >> 5;
    for (int r = wid; r < H_ * N_; r += 8) {
        int head = r >> 7, n = r & 127;
        const int* arow = Adj + ((size_t)b * N_ + n) * N_;
        float eb = eis[r];
        float ev[4]; int ok[4];
        float mx = -3.0e38f;
        #pragma unroll
        for (int it = 0; it < 4; ++it) {
            int m = lane + 32 * it;
            float e = eb + ejs[head * 128 + m];
            e = fmaxf(e, 0.01f * e);                // leaky_relu(0.01)
            ok[it] = arow[m] > 0;
            ev[it] = ok[it] ? e : -3.0e38f;
            mx = fmaxf(mx, ev[it]);
        }
        mx = wred_max(mx);
        float p[4]; float s = 0.f;
        #pragma unroll
        for (int it = 0; it < 4; ++it) {
            p[it] = ok[it] ? __expf(ev[it] - mx) : 0.f;
            s += p[it];
        }
        s = wred_sum(s);
        float accd[8] = {0.f,0.f,0.f,0.f,0.f,0.f,0.f,0.f};
        #pragma unroll
        for (int it = 0; it < 4; ++it) {
            int m = lane + 32 * it;
            const float* hv = &hs[head * 1024 + m * 8];
            float pm = p[it];
            #pragma unroll
            for (int d = 0; d < 8; ++d) accd[d] += pm * hv[d];
        }
        #pragma unroll
        for (int d = 0; d < 8; ++d) accd[d] = wred_sum(accd[d]);
        if (lane < 8)
            out[((size_t)b * N_ + n) * G_ + head * 8 + lane] = accd[lane] / s;
    }
}

// ---------------------------------------------------------------- LayerNorm over [N,G] (in place) + ReLU
__global__ __launch_bounds__(256)
void ln2d_relu_kernel(float* __restrict__ g, const float* __restrict__ gam,
                      const float* __restrict__ bet)
{
    __shared__ float r1[256], r2[256];
    const int b = blockIdx.x, tid = threadIdx.x;
    float* row = g + (size_t)b * 6144;
    float s = 0.f, q = 0.f;
    for (int i = tid; i < 6144; i += 256) { float x = row[i]; s += x; q += x * x; }
    r1[tid] = s; r2[tid] = q; __syncthreads();
    for (int st = 128; st > 0; st >>= 1) {
        if (tid < st) { r1[tid] += r1[tid + st]; r2[tid] += r2[tid + st]; }
        __syncthreads();
    }
    float m  = r1[0] * (1.f / 6144.f);
    float v  = r2[0] * (1.f / 6144.f) - m * m;
    float rs = rsqrtf(v + EPSV);
    for (int i = tid; i < 6144; i += 256) {
        float y = (row[i] - m) * rs * gam[i] + bet[i];
        row[i] = fmaxf(y, 0.f);
    }
}

// ---------------------------------------------------------------- residual (X@Wr+br) + LayerNorm + ReLU -> f16
__global__ __launch_bounds__(256)
void resid_ln_kernel(const float* __restrict__ g2raw, const float* __restrict__ X,
                     const float* __restrict__ Wr, const float* __restrict__ br,
                     const float* __restrict__ gam, const float* __restrict__ bet,
                     h16_t* __restrict__ out16)
{
    __shared__ float t[6144];
    __shared__ float Xs[N_ * F_];
    __shared__ float Wrs[F_ * G_];
    __shared__ float r1[256], r2[256];
    const int b = blockIdx.x, tid = threadIdx.x;
    for (int i = tid; i < N_ * F_; i += 256) Xs[i] = X[(size_t)b * (N_ * F_) + i];
    for (int i = tid; i < F_ * G_; i += 256) Wrs[i] = Wr[i];
    __syncthreads();
    float s = 0.f, q = 0.f;
    for (int i = tid; i < 6144; i += 256) {
        int n = i / 48, c = i - n * 48;
        float acc = g2raw[(size_t)b * 6144 + i] + br[c];
        const float* xr = &Xs[n * 16];
        #pragma unroll
        for (int f = 0; f < 16; ++f) acc += xr[f] * Wrs[f * 48 + c];
        t[i] = acc; s += acc; q += acc * acc;
    }
    r1[tid] = s; r2[tid] = q; __syncthreads();
    for (int st = 128; st > 0; st >>= 1) {
        if (tid < st) { r1[tid] += r1[tid + st]; r2[tid] += r2[tid + st]; }
        __syncthreads();
    }
    float m  = r1[0] * (1.f / 6144.f);
    float v  = r2[0] * (1.f / 6144.f) - m * m;
    float rs = rsqrtf(v + EPSV);
    for (int i = tid; i < 6144; i += 256) {
        float y = (t[i] - m) * rs * gam[i] + bet[i];
        out16[(size_t)b * 6144 + i] = (h16_t)fmaxf(y, 0.f);
    }
}

// ---------------------------------------------------------------- WMMA GEMM: C[M,N] = A[M,K] * W[N,K]^T (+bias)(+relu)
// A,W f16 row-major, f32 accumulate. K%32==0, M%64==0, N%64==0.
// Per-wave tile: 64 rows x 64 cols (4 A-frags x 4 B-frags -> 16 WMMA / K-step),
// which cuts weight-matrix re-reads 4x vs a 16-row tile (L2 traffic bound).
// ISA A-frag layout (16-bit A 16x32): lane<16 -> K {0..7,16..23}; lane>=16 -> K {8..15,24..31}
__device__ __forceinline__ v16h load_afrag(const h16_t* __restrict__ A, int lda,
                                           int row0, int k0, int lane) {
    int r = row0 + (lane & 15);
    int koff = k0 + ((lane >> 4) << 3);
    const h16_t* p = A + (size_t)r * lda + koff;
    union { v16h v; v8h h[2]; } u;
    u.h[0] = *(const v8h*)p;          // K = koff .. koff+7
    u.h[1] = *(const v8h*)(p + 16);   // K = koff+16 .. koff+23
    return u.v;
}
// B-frag layout (16-bit B 32x16), matching A: lane<16 -> col=lane, K {0..7,16..23}; lane>=16 -> col=lane-16, K {8..15,24..31}
__device__ __forceinline__ v16h load_bfrag(const h16_t* __restrict__ W, int ldw,
                                           int n0, int k0, int lane) {
    int c = n0 + (lane & 15);
    int koff = k0 + ((lane >> 4) << 3);
    const h16_t* p = W + (size_t)c * ldw + koff;
    union { v16h v; v8h h[2]; } u;
    u.h[0] = *(const v8h*)p;
    u.h[1] = *(const v8h*)(p + 16);
    return u.v;
}

__global__ __launch_bounds__(128)
void gemm_wmma_kernel(const h16_t* __restrict__ A, const h16_t* __restrict__ W,
                      const float* __restrict__ bias, float* __restrict__ Cf32,
                      h16_t* __restrict__ Cf16, int M, int N, int K, int relu)
{
    const int lane = threadIdx.x & 31;
    const int wave = threadIdx.x >> 5;
    const int row0 = blockIdx.x * 64;                  // 64 M-rows per block/wave
    const int n0base = (blockIdx.y * 4 + wave) * 64;   // 64 N-cols per wave
    if (row0 >= M || n0base >= N) return;

    const v8f zero = {0.f,0.f,0.f,0.f,0.f,0.f,0.f,0.f};
    v8f acc[4][4] = {{zero,zero,zero,zero},{zero,zero,zero,zero},
                     {zero,zero,zero,zero},{zero,zero,zero,zero}};  // [btile][atile]

    for (int k0 = 0; k0 < K; k0 += 32) {
        if (k0 + 32 < K) {  // uniform-branch prefetch of next A/B chunks
            __builtin_prefetch(A + (size_t)(row0 + (lane & 15)) * K + k0 + 32, 0, 1);
            __builtin_prefetch(W + (size_t)(n0base + (lane & 15)) * K + k0 + 32, 0, 1);
        }
        v16h a0 = load_afrag(A, K, row0 +  0, k0, lane);
        v16h a1 = load_afrag(A, K, row0 + 16, k0, lane);
        v16h a2 = load_afrag(A, K, row0 + 32, k0, lane);
        v16h a3 = load_afrag(A, K, row0 + 48, k0, lane);
        #pragma unroll
        for (int bt = 0; bt < 4; ++bt) {
            v16h b = load_bfrag(W, K, n0base + bt * 16, k0, lane);
            acc[bt][0] = __builtin_amdgcn_wmma_f32_16x16x32_f16(false, a0, false, b, (short)0, acc[bt][0], false, false);
            acc[bt][1] = __builtin_amdgcn_wmma_f32_16x16x32_f16(false, a1, false, b, (short)0, acc[bt][1], false, false);
            acc[bt][2] = __builtin_amdgcn_wmma_f32_16x16x32_f16(false, a2, false, b, (short)0, acc[bt][2], false, false);
            acc[bt][3] = __builtin_amdgcn_wmma_f32_16x16x32_f16(false, a3, false, b, (short)0, acc[bt][3], false, false);
        }
    }

    // C layout: lane<16 -> (M=v, N=lane); lane>=16 -> (M=8+v, N=lane-16)
    const int nlo = lane & 15;
    const int mhi = (lane >> 4) << 3;
    #pragma unroll
    for (int bt = 0; bt < 4; ++bt) {
        int n = n0base + bt * 16 + nlo;
        float bv = bias ? bias[n] : 0.f;
        #pragma unroll
        for (int at = 0; at < 4; ++at) {
            #pragma unroll
            for (int v = 0; v < 8; ++v) {
                int m = row0 + at * 16 + mhi + v;
                float val = acc[bt][at][v] + bv;
                if (relu) val = fmaxf(val, 0.f);
                if (Cf32) Cf32[(size_t)m * N + n] = val;
                if (Cf16) Cf16[(size_t)m * N + n] = (h16_t)val;
            }
        }
    }
}

// ---------------------------------------------------------------- GRU elementwise (hidden state == 0)
// gi:[B,768] = x@Wih^T + bih already; gh = bhh. h' = (1-z)*tanh(inn + r*hn)
__global__ void gru_elem_kernel(const float* __restrict__ gi, const float* __restrict__ bhh,
                                float* __restrict__ hf32, h16_t* __restrict__ hf16)
{
    int idx = blockIdx.x * 256 + threadIdx.x;
    if (idx >= B_ * GRU_) return;
    int b = idx >> 8, j = idx & 255;
    const float* g = gi + (size_t)b * 768;
    float r = 1.f / (1.f + __expf(-(g[j]       + bhh[j])));
    float z = 1.f / (1.f + __expf(-(g[256 + j] + bhh[256 + j])));
    float n = tanhf(g[512 + j] + r * bhh[512 + j]);
    float h = (1.f - z) * n;
    if (hf32) hf32[idx] = h;
    if (hf16) hf16[idx] = (h16_t)h;
}

// ---------------------------------------------------------------- LayerNorm over 256 (wave per row) -> f16
__global__ __launch_bounds__(256)
void ln1d_kernel(const float* __restrict__ h, const float* __restrict__ gam,
                 const float* __restrict__ bet, h16_t* __restrict__ out16)
{
    int lane = threadIdx.x & 31, wid = threadIdx.x >> 5;
    int row = blockIdx.x * 8 + wid;
    const float* x = h + (size_t)row * 256;
    float xv[8]; float s = 0.f, q = 0.f;
    #pragma unroll
    for (int i = 0; i < 8; ++i) {
        xv[i] = x[lane + 32 * i];
        s += xv[i]; q += xv[i] * xv[i];
    }
    s = wred_sum(s); q = wred_sum(q);
    float m  = s * (1.f / 256.f);
    float v  = q * (1.f / 256.f) - m * m;
    float rs = rsqrtf(v + EPSV);
    #pragma unroll
    for (int i = 0; i < 8; ++i) {
        int j = lane + 32 * i;
        out16[(size_t)row * 256 + j] = (h16_t)((xv[i] - m) * rs * gam[j] + bet[j]);
    }
}

// ---------------------------------------------------------------- launch
extern "C" void kernel_launch(void* const* d_in, const int* in_sizes, int n_in,
                              void* d_out, int out_size, void* d_ws, size_t ws_size,
                              hipStream_t stream)
{
    (void)in_sizes; (void)n_in; (void)out_size; (void)ws_size;

    const float* X    = (const float*)d_in[0];
    const int*   Adj  = (const int*)  d_in[1];
    const float* W1   = (const float*)d_in[2];
    const float* a1   = (const float*)d_in[3];
    const float* W2   = (const float*)d_in[4];
    const float* a2   = (const float*)d_in[5];
    const float* Wr   = (const float*)d_in[6];
    const float* br   = (const float*)d_in[7];
    const float* g1   = (const float*)d_in[8];
    const float* b1   = (const float*)d_in[9];
    const float* g2   = (const float*)d_in[10];
    const float* b2   = (const float*)d_in[11];
    const float* Wih0 = (const float*)d_in[12];
    const float* bih0 = (const float*)d_in[14];
    const float* bhh0 = (const float*)d_in[15];
    const float* Wih1 = (const float*)d_in[16];
    const float* bih1 = (const float*)d_in[18];
    const float* bhh1 = (const float*)d_in[19];
    const float* Wih2 = (const float*)d_in[20];
    const float* bih2 = (const float*)d_in[22];
    const float* bhh2 = (const float*)d_in[23];
    const float* gng  = (const float*)d_in[24];
    const float* gnb  = (const float*)d_in[25];
    const float* We   = (const float*)d_in[26];
    const float* be   = (const float*)d_in[27];
    const float* Wp   = (const float*)d_in[28];
    const float* bp   = (const float*)d_in[29];

    char* ws = (char*)d_ws;
    size_t off = 0;
    auto alloc = [&](size_t bytes) -> char* {
        char* p = ws + off;
        off += (bytes + 255) & ~(size_t)255;
        return p;
    };
    float* bufA    = (float*)alloc(sizeof(float) * (size_t)B_ * 6144);   // g1 raw / post-LN
    float* bufB    = (float*)alloc(sizeof(float) * (size_t)B_ * 6144);   // g2 raw
    h16_t* x16     = (h16_t*)alloc(sizeof(h16_t) * (size_t)B_ * 6144);   // GRU0 input
    h16_t* wih0_16 = (h16_t*)alloc(sizeof(h16_t) * (size_t)768 * 6144);
    h16_t* wih1_16 = (h16_t*)alloc(sizeof(h16_t) * (size_t)768 * 256);
    h16_t* wih2_16 = (h16_t*)alloc(sizeof(h16_t) * (size_t)768 * 256);
    h16_t* weT16   = (h16_t*)alloc(sizeof(h16_t) * (size_t)EMB_ * GRU_); // [128,256]
    h16_t* wpT16   = (h16_t*)alloc(sizeof(h16_t) * (size_t)N_ * EMB_);   // [128,128]
    float* gi      = (float*)alloc(sizeof(float) * (size_t)B_ * 768);
    h16_t* h1_16   = (h16_t*)alloc(sizeof(h16_t) * (size_t)B_ * GRU_);
    h16_t* h2_16   = (h16_t*)alloc(sizeof(h16_t) * (size_t)B_ * GRU_);
    float* h3_f32  = (float*)alloc(sizeof(float) * (size_t)B_ * GRU_);
    h16_t* h3ln16  = (h16_t*)alloc(sizeof(h16_t) * (size_t)B_ * GRU_);
    h16_t* emb16   = (h16_t*)alloc(sizeof(h16_t) * (size_t)B_ * EMB_);

    float* pred = (float*)d_out;                      // [1024,128]
    float* emb  = (float*)d_out + (size_t)B_ * EMB_;  // [1024,128]

    // weight conversion (deterministic, every call)
    cvt_f16_kernel <<<(768 * 6144 + 255) / 256, 256, 0, stream>>>(Wih0, wih0_16, 768 * 6144);
    cvt_f16_kernel <<<(768 * 256  + 255) / 256, 256, 0, stream>>>(Wih1, wih1_16, 768 * 256);
    cvt_f16_kernel <<<(768 * 256  + 255) / 256, 256, 0, stream>>>(Wih2, wih2_16, 768 * 256);
    cvtT_f16_kernel<<<(256 * 128  + 255) / 256, 256, 0, stream>>>(We, weT16, 256, 128);
    cvtT_f16_kernel<<<(128 * 128  + 255) / 256, 256, 0, stream>>>(Wp, wpT16, 128, 128);

    // GAT stack
    gat_kernel<16><<<B_, 256, 0, stream>>>(X, Adj, W1, a1, bufA);
    ln2d_relu_kernel<<<B_, 256, 0, stream>>>(bufA, g1, b1);
    gat_kernel<48><<<B_, 256, 0, stream>>>(bufA, Adj, W2, a2, bufB);
    resid_ln_kernel<<<B_, 256, 0, stream>>>(bufB, X, Wr, br, g2, b2, x16);

    // GRU stack (hidden state == 0 -> only input GEMMs)
    dim3 blk(128);
    gemm_wmma_kernel<<<dim3(16, 3), blk, 0, stream>>>(x16,   wih0_16, bih0, gi, nullptr, 1024, 768, 6144, 0);
    gru_elem_kernel <<<1024, 256, 0, stream>>>(gi, bhh0, nullptr, h1_16);
    gemm_wmma_kernel<<<dim3(16, 3), blk, 0, stream>>>(h1_16, wih1_16, bih1, gi, nullptr, 1024, 768, 256, 0);
    gru_elem_kernel <<<1024, 256, 0, stream>>>(gi, bhh1, nullptr, h2_16);
    gemm_wmma_kernel<<<dim3(16, 3), blk, 0, stream>>>(h2_16, wih2_16, bih2, gi, nullptr, 1024, 768, 256, 0);
    gru_elem_kernel <<<1024, 256, 0, stream>>>(gi, bhh2, h3_f32, nullptr);

    // head
    ln1d_kernel<<<128, 256, 0, stream>>>(h3_f32, gng, gnb, h3ln16);
    gemm_wmma_kernel<<<dim3(16, 1), blk, 0, stream>>>(h3ln16, weT16, be, emb,  emb16,   1024, 128, 256, 1);
    gemm_wmma_kernel<<<dim3(16, 1), blk, 0, stream>>>(emb16,  wpT16, bp, pred, nullptr, 1024, 128, 128, 0);
}
